// DSASparseAttentionWrapper_40372692583232
// MI455X (gfx1250) — matrix operand order repaired and checked by reading, not compiled, not run on hardware
//
#include <hip/hip_runtime.h>
#include <hip/hip_bf16.h>

typedef __attribute__((ext_vector_type(16))) _Float16 v16h;
typedef __attribute__((ext_vector_type(8)))  float    v8f;

// Problem dims (fixed by the reference harness)
static constexpr int Bn = 2, Hn = 8, Sn = 2048, Dn = 64, VDn = 64, Tn = 128;

// One workgroup per (b,s): 8 waves = 8 heads. Indices/indexer-scores are shared
// across heads, staged once in LDS. Each wave does the full sparse-attention
// pipeline for its head with v_wmma_f32_16x16x32_f16.
__global__ __launch_bounds__(256) void dsa_sparse_attn_kernel(
    const float* __restrict__ q, const float* __restrict__ k,
    const float* __restrict__ v, const int* __restrict__ topk_idx,
    const float* __restrict__ topk_sc, float* __restrict__ out)
{
    __shared__ int   sIdx[Tn];
    __shared__ float sTs[Tn];

    const int bs  = blockIdx.x;
    const int b   = bs / Sn;
    const int s   = bs % Sn;
    const int tid = threadIdx.x;

    // Cooperative staging: 128 indices + 128 indexer scores for this (b,s).
    if (tid < Tn) {
        sIdx[tid] = topk_idx[((size_t)b * Sn + s) * Tn + tid];
    } else if (tid < 2 * Tn) {
        const int t = tid - Tn;
        sTs[t] = topk_sc[((size_t)b * Sn + s) * Tn + t];
    }
    __syncthreads();

    const int h     = tid >> 5;        // wave id == head
    const int lane  = tid & 31;
    const int laneN = lane & 15;
    const int hi    = (lane >> 4) & 1; // 0 = lanes 0-15, 1 = lanes 16-31

    const size_t bh   = (size_t)(b * Hn + h);
    const float* qRow = q + (bh * Sn + s) * Dn;
    const float* kB   = k + bh * (size_t)Sn * Dn;
    const float* vB   = v + bh * (size_t)Sn * VDn;

    // ---- B fragment for QK: q broadcast into all 16 N columns ----------
    // B (32x16), contiguous k order used by this code: lane holds column N = lane%16;
    // K rows = (lane<16 ? 0..15 : 16..31), element e -> K row e (+16 if hi).
    v16h bq[2];
#pragma unroll
    for (int dc = 0; dc < 2; ++dc) {
        const int d0 = dc * 32 + hi * 16;
#pragma unroll
        for (int e = 0; e < 16; ++e) bq[dc][e] = (_Float16)qRow[d0 + e];
    }

    // ---- Phase 1: scores[t] = (q . K[idx[t]]) * scale -------------------
    // A (16x32), same contiguous k order as B: lane holds row M = lane%16,
// element e -> K(=d) = dBase + e, dBase = dc*32 + hi*16.
    const float scale = 0.125f; // 64^-0.5
    float wv[8][8];
    float mx = -3.402823466e38f;
#pragma unroll
    for (int c = 0; c < 8; ++c) {
        const int row  = sIdx[c * 16 + laneN];     // t-row = 16c + lane%16
        const float* kR = kB + (size_t)row * Dn;
        v8f acc = {};
#pragma unroll
        for (int dc = 0; dc < 2; ++dc) {
            const int d0 = dc * 32 + hi * 16;
            v16h a;
#pragma unroll
            for (int e = 0; e < 8; ++e) a[e]     = (_Float16)kR[d0 + e];
#pragma unroll
            for (int e = 0; e < 8; ++e) a[8 + e] = (_Float16)kR[d0 + 8 + e];
            acc = __builtin_amdgcn_wmma_f32_16x16x32_f16(
                false, a, false, bq[dc], (short)0, acc, false, false);
        }
        // C/D layout: lane holds N = lane%16 (all columns identical here);
        // VGPR j -> M = j + 8*hi  => this lane owns t = 16c + j + 8*hi.
#pragma unroll
        for (int j = 0; j < 8; ++j) {
            const float sc = acc[j] * scale;
            wv[c][j] = sc;
            mx = fmaxf(mx, sc);
        }
    }
    // Lanes L and L^16 together cover all 128 t's -> one xor-16 reduction.
    mx = fmaxf(mx, __shfl_xor(mx, 16, 32));

    // softmax * indexer score, renormalized (softmax denom cancels):
    // w[t] = exp(s-m)*ts[t] / (sum exp(s-m)*ts + eps)
    float sum = 0.f;
#pragma unroll
    for (int c = 0; c < 8; ++c) {
#pragma unroll
        for (int j = 0; j < 8; ++j) {
            const float p = __expf(wv[c][j] - mx) * sTs[c * 16 + j + hi * 8];
            wv[c][j] = p;
            sum += p;
        }
    }
    sum += __shfl_xor(sum, 16, 32);
    const float inv = 1.f / (sum + 1e-12f);

    // ---- Phase 2: out[d] = sum_t w[t] * V[idx[t], d] --------------------
    // A = weights broadcast across all 16 M rows. The A K-striping lines up
    // exactly with where this lane stored its scores: elements 0..7 come
    // from chunk 2*c2, elements 8..15 from chunk 2*c2+1 (both halves).
    v8f oacc[4] = {{}, {}, {}, {}};
#pragma unroll
    for (int c2 = 0; c2 < 4; ++c2) {
        v16h aw;
#pragma unroll
        for (int e = 0; e < 8; ++e) aw[e]     = (_Float16)(wv[2 * c2][e] * inv);
#pragma unroll
        for (int e = 0; e < 8; ++e) aw[8 + e] = (_Float16)(wv[2 * c2 + 1][e] * inv);

        int rows[16];
#pragma unroll
        for (int e = 0; e < 16; ++e) rows[e] = sIdx[c2 * 32 + hi * 8 + e + (e < 8 ? 0 : 8)];

#pragma unroll
        for (int dt = 0; dt < 4; ++dt) {
            // B = V-gather tile: lane column n = lane%16, element e -> t row.
            // For fixed e, lanes 0-15 read 64 contiguous bytes of one V row.
            v16h bv;
#pragma unroll
            for (int e = 0; e < 16; ++e)
                bv[e] = (_Float16)vB[(size_t)rows[e] * VDn + dt * 16 + laneN];
            oacc[dt] = __builtin_amdgcn_wmma_f32_16x16x32_f16(
                false, aw, false, bv, (short)0, oacc[dt], false, false);
        }
    }

    // All M rows of D identical (A rows broadcast); column n lives in lane n.
    if (lane < 16) {
        float* oRow = out + (bh * Sn + s) * VDn;
#pragma unroll
        for (int dt = 0; dt < 4; ++dt) oRow[dt * 16 + lane] = oacc[dt][0];
    }
}

extern "C" void kernel_launch(void* const* d_in, const int* in_sizes, int n_in,
                              void* d_out, int out_size, void* d_ws, size_t ws_size,
                              hipStream_t stream) {
    const float* q  = (const float*)d_in[0];
    const float* k  = (const float*)d_in[1];
    const float* v  = (const float*)d_in[2];
    const int*   ti = (const int*)d_in[3];
    const float* ts = (const float*)d_in[4];
    float* out      = (float*)d_out;

    dim3 grid(Bn * Sn);   // one workgroup per (b,s)
    dim3 block(256);      // 8 waves = 8 heads
    hipLaunchKernelGGL(dsa_sparse_attn_kernel, grid, block, 0, stream,
                       q, k, v, ti, ts, out);
}
